// Attention_47553877901998
// MI455X (gfx1250) — matrix operand order repaired and checked by reading, not compiled, not run on hardware
//
#include <hip/hip_runtime.h>
#include <hip/hip_bf16.h>

typedef __attribute__((ext_vector_type(16))) _Float16 v16h;
typedef __attribute__((ext_vector_type(8)))  float    v8f;

#define NEG_BIG (-3.402823466e38f)

// ---- fragment loaders (wave32 WMMA layouts, 16B-aligned LDS/global reads) ----
__device__ __forceinline__ v16h ld_frag_2x8(const _Float16* p0, const _Float16* p1) {
    v16h f;
    uint4* d = reinterpret_cast<uint4*>(&f);
    d[0] = *reinterpret_cast<const uint4*>(p0);
    d[1] = *reinterpret_cast<const uint4*>(p1);
    return f;
}
__device__ __forceinline__ v16h ld_frag_16(const _Float16* p) {
    v16h f;
    uint4* d = reinterpret_cast<uint4*>(&f);
    const uint4* s = reinterpret_cast<const uint4*>(p);
    d[0] = s[0]; d[1] = s[1];
    return f;
}

// ---- CDNA5 async global->LDS copy (ASYNCcnt-tracked, bypasses VGPRs) ----
__device__ __forceinline__ void async_copy_b128(unsigned lds_addr, const void* gaddr) {
    asm volatile("global_load_async_to_lds_b128 %0, %1, off"
                 :: "v"(lds_addr), "v"(gaddr) : "memory");
}
__device__ __forceinline__ void wait_async0() {
    asm volatile("s_wait_asynccnt 0x0" ::: "memory");
}

// =====================  f32 -> f16 cast  =====================
__global__ void cast_f32_f16(const float* __restrict__ in, _Float16* __restrict__ out, int n) {
    int i = blockIdx.x * blockDim.x + threadIdx.x;
    if (i < n) out[i] = (_Float16)in[i];
}

// =====================  RMSNorm + cast (row = token, D=1024)  =====================
__global__ void __launch_bounds__(256)
rmsnorm_cast(const float* __restrict__ X, const float* __restrict__ w, _Float16* __restrict__ out) {
    __shared__ float red[8];
    const int row = blockIdx.x;              // 4096 rows
    const float* x = X + (size_t)row * 1024;
    float ss = 0.f;
#pragma unroll
    for (int i = 0; i < 4; ++i) {
        float v = x[threadIdx.x + i * 256];
        ss += v * v;
    }
#pragma unroll
    for (int ofs = 1; ofs < 32; ofs <<= 1) ss += __shfl_xor(ss, ofs, 32);
    if ((threadIdx.x & 31) == 0) red[threadIdx.x >> 5] = ss;
    __syncthreads();
    float tot = 0.f;
#pragma unroll
    for (int i = 0; i < 8; ++i) tot += red[i];
    const float scale = rsqrtf(tot * (1.0f / 1024.0f) + 1.1920929e-07f);
#pragma unroll
    for (int i = 0; i < 4; ++i) {
        int c = threadIdx.x + i * 256;
        out[(size_t)row * 1024 + c] = (_Float16)(x[c] * scale * w[c]);
    }
}

// =====================  Generic WMMA GEMM: C[M,N] = A[M,K] * B[K,N]  =====================
// block tile 128x64, BK=32, 256 threads = 8 waves; wave w -> rows [w*16, w*16+16)
__global__ void __launch_bounds__(256)
gemm_f16f32(const _Float16* __restrict__ A, const _Float16* __restrict__ Bw,
            float* __restrict__ C, int M, int N, int K) {
    __shared__ _Float16 As[128 * 40];   // [128][32] pad->40
    __shared__ _Float16 Bt[64 * 40];    // transposed [n=64][k=32] pad->40
    const int tid  = threadIdx.x;
    const int lane = tid & 31;
    const int w    = tid >> 5;
    const int Mb   = blockIdx.y * 128;
    const int Nb   = blockIdx.x * 64;
    v8f acc[4] = {};
    const int ar = tid >> 1, ac = (tid & 1) * 16;   // A tile loader: 16 halves each
    const int br = tid >> 3, bc = (tid & 7) * 8;    // B tile loader: 8 halves each
    const int m  = lane & 15, hi = lane >> 4;
    const unsigned ldsA = (unsigned)(uintptr_t)&As[ar * 40 + ac];

    for (int k0 = 0; k0 < K; k0 += 32) {
        __syncthreads();
        {   // A tile 128x32: async DMA straight into LDS (no VGPR staging)
            const _Float16* ga = A + (size_t)(Mb + ar) * K + k0 + ac;
            async_copy_b128(ldsA, ga);
            async_copy_b128(ldsA + 16, ga + 8);
        }
        {   // B tile 32x64, stored transposed for contiguous fragment reads
            const uint4 vb = *reinterpret_cast<const uint4*>(Bw + (size_t)(k0 + br) * N + Nb + bc);
            const _Float16* hb = reinterpret_cast<const _Float16*>(&vb);
#pragma unroll
            for (int i = 0; i < 8; ++i) Bt[(bc + i) * 40 + br] = hb[i];
        }
        if (k0 + 32 < K)
            __builtin_prefetch(A + (size_t)(Mb + ar) * K + k0 + 32 + ac, 0, 1);
        wait_async0();
        __syncthreads();
        // A frag: lane m = row, lane-group picks contiguous K half {16*hi..16*hi+15}
        v16h af = ld_frag_16(&As[(w * 16 + m) * 40 + hi * 16]);
        // issue all B-fragment loads first, then back-to-back WMMAs
        v16h bf[4];
#pragma unroll
        for (int t = 0; t < 4; ++t)
            bf[t] = ld_frag_16(&Bt[(t * 16 + m) * 40 + hi * 16]);
#pragma unroll
        for (int t = 0; t < 4; ++t)
            acc[t] = __builtin_amdgcn_wmma_f32_16x16x32_f16(false, af, false, bf[t],
                                                            (short)0, acc[t], false, false);
    }
#pragma unroll
    for (int t = 0; t < 4; ++t)
#pragma unroll
        for (int r = 0; r < 8; ++r)
            C[(size_t)(Mb + w * 16 + hi * 8 + r) * N + Nb + t * 16 + m] = acc[t][r];
}

// =====================  per-head q norm: l2norm * (gamma+1)*sqrt(64)  =====================
// grid (B*N, QH), 32 threads; out layout [b, qh, n, dh]
__global__ void qnorm_kernel(const float* __restrict__ P, const float* __restrict__ gamma,
                             _Float16* __restrict__ out) {
    const int tok = blockIdx.x, qh = blockIdx.y, lane = threadIdx.x;
    const int b = tok >> 11, i = tok & 2047;
    const float* p = P + (size_t)tok * 1024 + qh * 64;
    float v0 = p[lane], v1 = p[lane + 32];
    float ss = v0 * v0 + v1 * v1;
#pragma unroll
    for (int ofs = 1; ofs < 32; ofs <<= 1) ss += __shfl_xor(ss, ofs, 32);
    const float inv = 1.0f / fmaxf(sqrtf(ss), 1e-12f);
    _Float16* o = out + (((size_t)b * 16 + qh) * 2048 + i) * 64;
    o[lane]      = (_Float16)(v0 * inv * (gamma[qh * 64 + lane] + 1.0f) * 8.0f);
    o[lane + 32] = (_Float16)(v1 * inv * (gamma[qh * 64 + lane + 32] + 1.0f) * 8.0f);
}

// k norm + v cast; kv rows are [k(0:512) | v(512:1024)]; grid (B*N, H), 32 threads
__global__ void kvnorm_kernel(const float* __restrict__ P, const float* __restrict__ gamma,
                              _Float16* __restrict__ kout, _Float16* __restrict__ vout) {
    const int tok = blockIdx.x, h = blockIdx.y, lane = threadIdx.x;
    const int b = tok >> 11, i = tok & 2047;
    const float* pk = P + (size_t)tok * 1024 + h * 64;
    const float* pv = pk + 512;
    float v0 = pk[lane], v1 = pk[lane + 32];
    float ss = v0 * v0 + v1 * v1;
#pragma unroll
    for (int ofs = 1; ofs < 32; ofs <<= 1) ss += __shfl_xor(ss, ofs, 32);
    const float inv = 1.0f / fmaxf(sqrtf(ss), 1e-12f);
    const size_t base = (((size_t)b * 8 + h) * 2048 + i) * 64;
    kout[base + lane]      = (_Float16)(v0 * inv * (gamma[h * 64 + lane] + 1.0f) * 8.0f);
    kout[base + lane + 32] = (_Float16)(v1 * inv * (gamma[h * 64 + lane + 32] + 1.0f) * 8.0f);
    vout[base + lane]      = (_Float16)pv[lane];
    vout[base + lane + 32] = (_Float16)pv[lane + 32];
}

// =====================  flash attention (softcap-before-scale, causal)  =====================
// grid (N/64, QH, B); 128 threads = 4 waves; wave w -> 16 query rows
__global__ void __launch_bounds__(128)
attn_fwd(const _Float16* __restrict__ Q, const _Float16* __restrict__ Kg,
         const _Float16* __restrict__ Vg, float* __restrict__ O) {
    __shared__ _Float16 Ks[64 * 72];      // [j][dh]  (== K^T B-operand, row-major)
    __shared__ _Float16 Vs[64 * 72];      // transposed [dh][j]
    __shared__ _Float16 Ps[4][16 * 72];   // per-wave exp(S) tile, A-operand layout
    const int tid = threadIdx.x, lane = tid & 31, w = tid >> 5;
    const int i0 = blockIdx.x * 64;
    const int qh = blockIdx.y;
    const int b  = blockIdx.z;
    const int h  = qh >> 1;               // GQA: 2 query heads per kv head
    const int m  = lane & 15, hi = lane >> 4;

    const _Float16* Qb = Q + (((size_t)b * 16 + qh) * 2048 + i0 + w * 16) * 64;
    const _Float16* Kb = Kg + (((size_t)b * 8 + h) * 2048) * 64;
    const _Float16* Vb = Vg + (((size_t)b * 8 + h) * 2048) * 64;

    v16h qf[2];
#pragma unroll
    for (int kc = 0; kc < 2; ++kc)
        qf[kc] = ld_frag_16(Qb + m * 64 + kc * 32 + hi * 16);

    float rowM[8], rowL[8];
    v8f o[4] = {};
#pragma unroll
    for (int r = 0; r < 8; ++r) { rowM[r] = NEG_BIG; rowL[r] = 0.f; }

    const int nJT = blockIdx.x + 1;           // causal: tiles with j0 <= i0
    const int lr = tid >> 1, lc = (tid & 1) * 32;
    const unsigned ldsK = (unsigned)(uintptr_t)&Ks[lr * 72 + lc];

    for (int jt = 0; jt < nJT; ++jt) {
        const int j0 = jt * 64;
        __syncthreads();
        {   // K tile [64][64] row-major: async DMA straight into LDS
            const _Float16* gk = Kb + (size_t)(j0 + lr) * 64 + lc;
#pragma unroll
            for (int c = 0; c < 4; ++c)
                async_copy_b128(ldsK + c * 16, gk + c * 8);
        }
        {   // V tile, transposed into [dh][j] (manual: async copy cannot transpose)
            const uint4* src = reinterpret_cast<const uint4*>(Vb + (size_t)(j0 + lr) * 64 + lc);
            uint4 tmp[4];
            tmp[0] = src[0]; tmp[1] = src[1]; tmp[2] = src[2]; tmp[3] = src[3];
            const _Float16* hv = reinterpret_cast<const _Float16*>(tmp);
#pragma unroll
            for (int i = 0; i < 32; ++i) Vs[(lc + i) * 72 + lr] = hv[i];
        }
        wait_async0();
        __syncthreads();

        // ---- S = Q * K^T  (16 x 64 per wave): all frag loads, then WMMA chain ----
        v16h kf[8];
#pragma unroll
        for (int t = 0; t < 4; ++t)
#pragma unroll
            for (int kc = 0; kc < 2; ++kc)
                kf[t * 2 + kc] = ld_frag_16(&Ks[(t * 16 + m) * 72 + kc * 32 + hi * 16]);
        v8f s[4] = {};
#pragma unroll
        for (int t = 0; t < 4; ++t)
#pragma unroll
            for (int kc = 0; kc < 2; ++kc)
                s[t] = __builtin_amdgcn_wmma_f32_16x16x32_f16(false, qf[kc], false, kf[t * 2 + kc],
                                                              (short)0, s[t], false, false);

        // ---- softcap (before scale), scale, causal mask ----
        float pbuf[4][8], rmax[8];
#pragma unroll
        for (int r = 0; r < 8; ++r) rmax[r] = NEG_BIG;
#pragma unroll
        for (int t = 0; t < 4; ++t) {
            const int jg = j0 + t * 16 + m;
#pragma unroll
            for (int r = 0; r < 8; ++r) {
                const int ig = i0 + w * 16 + hi * 8 + r;
                float v = tanhf(s[t][r] * 0.02f) * (50.0f * 0.125f);
                v = (jg <= ig) ? v : NEG_BIG;
                pbuf[t][r] = v;
                rmax[r] = fmaxf(rmax[r], v);
            }
        }
        // row max across the 16-lane column group
#pragma unroll
        for (int r = 0; r < 8; ++r) {
            float v = rmax[r];
            v = fmaxf(v, __shfl_xor(v, 1, 32));
            v = fmaxf(v, __shfl_xor(v, 2, 32));
            v = fmaxf(v, __shfl_xor(v, 4, 32));
            v = fmaxf(v, __shfl_xor(v, 8, 32));
            rmax[r] = v;
        }
        float alpha[8], rsum[8];
#pragma unroll
        for (int r = 0; r < 8; ++r) {
            const float mnew = fmaxf(rowM[r], rmax[r]);
            alpha[r] = __expf(rowM[r] - mnew);
            rowM[r]  = mnew;
            rsum[r]  = 0.f;
        }
        // p = exp(s - m_new): write per-wave P tile (A-operand layout source)
#pragma unroll
        for (int t = 0; t < 4; ++t)
#pragma unroll
            for (int r = 0; r < 8; ++r) {
                const float p = __expf(pbuf[t][r] - rowM[r]);
                rsum[r] += p;
                Ps[w][(hi * 8 + r) * 72 + t * 16 + m] = (_Float16)p;
            }
#pragma unroll
        for (int r = 0; r < 8; ++r) {
            float v = rsum[r];
            v += __shfl_xor(v, 1, 32);
            v += __shfl_xor(v, 2, 32);
            v += __shfl_xor(v, 4, 32);
            v += __shfl_xor(v, 8, 32);
            rowL[r] = rowL[r] * alpha[r] + v;
        }
#pragma unroll
        for (int t = 0; t < 4; ++t)
#pragma unroll
            for (int r = 0; r < 8; ++r)
                o[t][r] *= alpha[r];
        __syncthreads();

        // ---- O += P * V: all frag loads, then WMMA chain ----
        v16h pf[2];
#pragma unroll
        for (int kc = 0; kc < 2; ++kc)
            pf[kc] = ld_frag_16(&Ps[w][m * 72 + kc * 32 + hi * 16]);
        v16h vf[8];
#pragma unroll
        for (int t = 0; t < 4; ++t)
#pragma unroll
            for (int kc = 0; kc < 2; ++kc)
                vf[t * 2 + kc] = ld_frag_16(&Vs[(t * 16 + m) * 72 + kc * 32 + hi * 16]);
#pragma unroll
        for (int t = 0; t < 4; ++t)
#pragma unroll
            for (int kc = 0; kc < 2; ++kc)
                o[t] = __builtin_amdgcn_wmma_f32_16x16x32_f16(false, pf[kc], false, vf[t * 2 + kc],
                                                              (short)0, o[t], false, false);
    }

    // epilogue: normalize, store fp32 per query head
#pragma unroll
    for (int t = 0; t < 4; ++t)
#pragma unroll
        for (int r = 0; r < 8; ++r) {
            const float inv = 1.0f / rowL[r];
            O[(((size_t)b * 16 + qh) * 2048 + i0 + w * 16 + hi * 8 + r) * 64 + t * 16 + m]
                = o[t][r] * inv;
        }
}

// =====================  sum over group axis (g=2) -> [B*N, 512] f16  =====================
__global__ void combine_heads(const float* __restrict__ O, _Float16* __restrict__ out) {
    const int idx = blockIdx.x * blockDim.x + threadIdx.x;   // B*N*H*DH = 2^21
    if (idx >= 2 * 2048 * 8 * 64) return;
    const int dh = idx & 63;
    const int h  = (idx >> 6) & 7;
    const int i  = (idx >> 9) & 2047;
    const int b  = idx >> 20;
    const size_t o0 = (((size_t)b * 16 + 2 * h) * 2048 + i) * 64 + dh;
    const size_t o1 = (((size_t)b * 16 + 2 * h + 1) * 2048 + i) * 64 + dh;
    out[((size_t)(b * 2048 + i)) * 512 + h * 64 + dh] = (_Float16)(O[o0] + O[o1]);
}

// =====================  launcher  =====================
extern "C" void kernel_launch(void* const* d_in, const int* in_sizes, int n_in,
                              void* d_out, int out_size, void* d_ws, size_t ws_size,
                              hipStream_t stream) {
    const float* tokens  = (const float*)d_in[0];
    const float* norm_w  = (const float*)d_in[1];
    const float* Wq      = (const float*)d_in[2];
    const float* Wkv     = (const float*)d_in[3];
    const float* Wout    = (const float*)d_in[4];
    const float* q_gamma = (const float*)d_in[5];
    const float* k_gamma = (const float*)d_in[6];
    float* out = (float*)d_out;

    char* ws = (char*)d_ws;
    size_t off = 0;
    auto carve = [&](size_t bytes) {
        void* p = ws + off;
        off = (off + bytes + 255) & ~(size_t)255;
        return p;
    };
    _Float16* xh    = (_Float16*)carve((size_t)4096 * 1024 * 2);  // rmsnorm(x) f16
    _Float16* wqh   = (_Float16*)carve((size_t)1024 * 1024 * 2);
    _Float16* wkvh  = (_Float16*)carve((size_t)1024 * 1024 * 2);
    _Float16* wouth = (_Float16*)carve((size_t)512 * 1024 * 2);
    float*    qproj = (float*)carve((size_t)4096 * 1024 * 4);
    float*    kvprj = (float*)carve((size_t)4096 * 1024 * 4);
    _Float16* qh16  = (_Float16*)carve((size_t)2 * 16 * 2048 * 64 * 2);
    _Float16* kh16  = (_Float16*)carve((size_t)2 * 8 * 2048 * 64 * 2);
    _Float16* vh16  = (_Float16*)carve((size_t)2 * 8 * 2048 * 64 * 2);
    float*    Obuf  = (float*)carve((size_t)2 * 16 * 2048 * 64 * 4);
    _Float16* attnc = (_Float16*)carve((size_t)2 * 2048 * 512 * 2);

    // weights -> f16
    cast_f32_f16<<<(1048576 + 255) / 256, 256, 0, stream>>>(Wq, wqh, 1048576);
    cast_f32_f16<<<(1048576 + 255) / 256, 256, 0, stream>>>(Wkv, wkvh, 1048576);
    cast_f32_f16<<<(524288 + 255) / 256, 256, 0, stream>>>(Wout, wouth, 524288);
    // rmsnorm
    rmsnorm_cast<<<4096, 256, 0, stream>>>(tokens, norm_w, xh);
    // projections
    gemm_f16f32<<<dim3(1024 / 64, 4096 / 128), 256, 0, stream>>>(xh, wqh, qproj, 4096, 1024, 1024);
    gemm_f16f32<<<dim3(1024 / 64, 4096 / 128), 256, 0, stream>>>(xh, wkvh, kvprj, 4096, 1024, 1024);
    // per-head norms (+ layout transpose to [b, head, n, dh])
    qnorm_kernel<<<dim3(4096, 16), 32, 0, stream>>>(qproj, q_gamma, qh16);
    kvnorm_kernel<<<dim3(4096, 8), 32, 0, stream>>>(kvprj, k_gamma, kh16, vh16);
    // flash attention
    attn_fwd<<<dim3(2048 / 64, 16, 2), 128, 0, stream>>>(qh16, kh16, vh16, Obuf);
    // group-sum + cast
    combine_heads<<<(2097152 + 255) / 256, 256, 0, stream>>>(Obuf, attnc);
    // output projection -> fp32 result
    gemm_f16f32<<<dim3(1024 / 64, 4096 / 128), 256, 0, stream>>>(attnc, wouth, out, 4096, 1024, 512);
}
